// HierarchicalAttention_66803921322325
// MI455X (gfx1250) — matrix operand order repaired and checked by reading, not compiled, not run on hardware
//
#include <hip/hip_runtime.h>
#include <hip/hip_bf16.h>

// ---------------------------------------------------------------------------
// Hierarchical attention on MI455X (gfx1250), bf16 WMMA everywhere.
//   B=2, T=8200 (=8*1025), D_MODEL=1024, H=16, DK=DV=64, C=8 chunks,
//   NCHUNK=1024 local tokens/chunk, BLK=128 sliding-window block.
// ---------------------------------------------------------------------------

typedef __bf16 bf16;
typedef __attribute__((ext_vector_type(16))) __bf16 bf16x16;
typedef __attribute__((ext_vector_type(8)))  __bf16 bf16x8;
typedef __attribute__((ext_vector_type(8)))  float  f32x8;

__device__ __forceinline__ f32x8 wmma_bf16(bf16x16 a, bf16x16 b, f32x8 c) {
  // (neg_a, A, neg_b, B, c_mod, C, reuse_a, reuse_b)
  return __builtin_amdgcn_wmma_f32_16x16x32_bf16(false, a, false, b, (short)0, c,
                                                 false, false);
}

// A-fragment (16x32 bf16): per lane two 16B loads with a 16-element gap.
// Caller passes p already offset by (lane>=16 ? 8 : 0).
__device__ __forceinline__ bf16x16 load_frag_a(const bf16* p) {
  bf16x8 lo = *(const bf16x8*)(p);
  bf16x8 hi = *(const bf16x8*)(p + 16);
  bf16x16 r;
#pragma unroll
  for (int i = 0; i < 8; ++i) { r[i] = lo[i]; r[i + 8] = hi[i]; }
  return r;
}

// B-fragment (32x16 bf16): per lane two 16B loads with a 16-element gap.
// Caller passes p already offset by (lane>=16 ? 8 : 0).
__device__ __forceinline__ bf16x16 load_frag_b(const bf16* p) {
  bf16x8 lo = *(const bf16x8*)(p);
  bf16x8 hi = *(const bf16x8*)(p + 16);
  bf16x16 r;
#pragma unroll
  for (int i = 0; i < 8; ++i) { r[i] = lo[i]; r[i + 8] = hi[i]; }
  return r;
}

__device__ __forceinline__ float rmax16(float v) {
#pragma unroll
  for (int m = 8; m >= 1; m >>= 1) v = fmaxf(v, __shfl_xor(v, m, 16));
  return v;
}
__device__ __forceinline__ float rsum16(float v) {
#pragma unroll
  for (int m = 8; m >= 1; m >>= 1) v += __shfl_xor(v, m, 16);
  return v;
}

// ---------------------------------------------------------------------------
// fp32 -> bf16 flat convert (vectorized: 8 elements / lane)
// ---------------------------------------------------------------------------
__global__ __launch_bounds__(256) void k_cvt(const float* __restrict__ src,
                                             bf16* __restrict__ dst) {
  int i = blockIdx.x * 256 + threadIdx.x;          // 2,099,200 threads
  const float4* s = (const float4*)src + (size_t)i * 2;
  float4 x0 = s[0], x1 = s[1];
  bf16x8 o;
  o[0] = (bf16)x0.x; o[1] = (bf16)x0.y; o[2] = (bf16)x0.z; o[3] = (bf16)x0.w;
  o[4] = (bf16)x1.x; o[5] = (bf16)x1.y; o[6] = (bf16)x1.z; o[7] = (bf16)x1.w;
  *((bf16x8*)dst + i) = o;
}

// fp32 [1024x1024] -> bf16 transposed [N][K], LDS-tiled 32x32.
__global__ __launch_bounds__(256) void k_cvt_t(const float* __restrict__ W,
                                               bf16* __restrict__ Wt) {
  __shared__ float tile[32][33];
  int tx = threadIdx.x & 31, ty = threadIdx.x >> 5;   // 32 x 8
  int bx = blockIdx.x * 32;                            // n base
  int by = blockIdx.y * 32;                            // k base
#pragma unroll
  for (int r = ty; r < 32; r += 8)
    tile[r][tx] = W[(size_t)(by + r) * 1024 + bx + tx];
  __syncthreads();
#pragma unroll
  for (int r = ty; r < 32; r += 8)
    Wt[(size_t)(bx + r) * 1024 + by + tx] = (bf16)tile[tx][r];
}

// ---------------------------------------------------------------------------
// QKV projection: [16400 x 1024] x [1024 x 3072].
// One wave computes a 32x64 tile; ping-pong double-buffered over K so no
// register rotation copies are emitted. 513 M-groups x 48 N-strips.
// ---------------------------------------------------------------------------
__global__ __launch_bounds__(128) void k_qkv(const bf16* __restrict__ xb,
                                             const bf16* __restrict__ WqT,
                                             const bf16* __restrict__ WkT,
                                             const bf16* __restrict__ WvT,
                                             bf16* __restrict__ Qb,
                                             bf16* __restrict__ Kb,
                                             bf16* __restrict__ Vtl,
                                             float* __restrict__ gV) {
  int wave = blockIdx.x * 4 + (threadIdx.x >> 5);
  int lane = threadIdx.x & 31;
  int lrow = lane & 15, lhalf = lane >> 4;
  int mgrp = wave / 48;                   // 0..512
  int nstrip = wave - mgrp * 48;          // 0..47
  int m0 = mgrp * 32;
  int n0 = nstrip * 64;
  int which = n0 >> 10;                   // 0=Q 1=K 2=V
  int ncol0 = n0 & 1023;
  const bf16* Wt = (which == 0) ? WqT : (which == 1) ? WkT : WvT;

  int r0 = m0 + lrow;
  int r1 = (m0 + 16 + lrow < 16400) ? (m0 + 16 + lrow) : r0;  // clamp tail
  const bf16* arow0 = xb + (size_t)r0 * 1024;
  const bf16* arow1 = xb + (size_t)r1 * 1024;
  const bf16* brow[4];
#pragma unroll
  for (int nt = 0; nt < 4; ++nt)
    brow[nt] = Wt + (size_t)(ncol0 + nt * 16 + lrow) * 1024;

  f32x8 acc[8];
#pragma unroll
  for (int i = 0; i < 8; ++i) acc[i] = (f32x8)(0.0f);

  auto loadS = [&](int k, bf16x16& a0, bf16x16& a1, bf16x16* bb) {
    a0 = load_frag_a(arow0 + k + lhalf * 8);
    a1 = load_frag_a(arow1 + k + lhalf * 8);
#pragma unroll
    for (int nt = 0; nt < 4; ++nt)
      bb[nt] = load_frag_b(brow[nt] + k + lhalf * 8);
  };
  auto compS = [&](const bf16x16& a0, const bf16x16& a1, const bf16x16* bb) {
#pragma unroll
    for (int nt = 0; nt < 4; ++nt) {
      acc[nt * 2 + 0] = wmma_bf16(a0, bb[nt], acc[nt * 2 + 0]);
      acc[nt * 2 + 1] = wmma_bf16(a1, bb[nt], acc[nt * 2 + 1]);
    }
  };

  bf16x16 a0A, a1A, bA[4], a0B, a1B, bB[4];
  loadS(0, a0A, a1A, bA);
  for (int k = 32; k < 992; k += 64) {    // k = 32, 96, ..., 928 (15 iters)
    loadS(k, a0B, a1B, bB);
    compS(a0A, a1A, bA);
    loadS(k + 32, a0A, a1A, bA);
    compS(a0B, a1B, bB);
  }
  loadS(992, a0B, a1B, bB);
  compS(a0A, a1A, bA);
  compS(a0B, a1B, bB);

#pragma unroll
  for (int mi = 0; mi < 2; ++mi) {
    if (m0 + mi * 16 >= 16400) break;     // uniform tail guard
#pragma unroll
    for (int nt = 0; nt < 4; ++nt) {
      int ncol = ncol0 + nt * 16 + lrow;
      int h = ncol >> 6, d = ncol & 63;
#pragma unroll
      for (int j = 0; j < 8; ++j) {
        int m = m0 + mi * 16 + j + 8 * lhalf;
        int bidx = (m >= 8200) ? 1 : 0;
        int tt = m - bidx * 8200;
        float v = acc[nt * 2 + mi][j];
        if (which == 0) {
          Qb[(((size_t)(bidx * 16 + h)) * 8200 + tt) * 64 + d] = (bf16)v;
        } else if (which == 1) {
          Kb[(((size_t)(bidx * 16 + h)) * 8200 + tt) * 64 + d] = (bf16)v;
        } else {
          int c = tt / 1025;
          int r = tt - c * 1025;
          size_t bhc = ((size_t)(bidx * 16 + h)) * 8 + c;
          if (r == 0) gV[bhc * 64 + d] = v;                      // global token
          else Vtl[(bhc * 64 + d) * 1024 + (r - 1)] = (bf16)v;   // transposed
        }
      }
    }
  }
}

// ---------------------------------------------------------------------------
// Local sliding-window attention, flash-style. One wave = 16 queries of one
// (b,h,chunk). Window = [max(0,(qblk-1)*128), min(1024,(qblk+2)*128)); the
// reference's phantom edge blocks are skipped (softmax weight exactly zero).
// K fragments ping-pong double-buffered (block count is 8 or 12, always even);
// V fragments issued early so they overlap the softmax VALU work.
// ---------------------------------------------------------------------------
__global__ __launch_bounds__(128) void k_local(const bf16* __restrict__ Qb,
                                               const bf16* __restrict__ Kb,
                                               const bf16* __restrict__ Vtl,
                                               const float* __restrict__ gV,
                                               bf16* __restrict__ att) {
  __shared__ __align__(16) bf16 Plds[4][16 * 32];   // 1KB per wave
  int wv = threadIdx.x >> 5, lane = threadIdx.x & 31;
  int lrow = lane & 15, lhalf = lane >> 4;
  int gid = blockIdx.x * 4 + wv;                    // 0..16383
  int qt = gid & 63;
  int c = (gid >> 6) & 7;
  int h = (gid >> 9) & 15;
  int b = gid >> 13;
  size_t bh = (size_t)b * 16 + h;
  int tbase = c * 1025 + 1;                         // first local token

  const bf16* Qrow = Qb + (bh * 8200 + tbase + qt * 16 + lrow) * 64;
  bf16x16 aq0 = load_frag_a(Qrow + lhalf * 8);        // dk 0..31
  bf16x16 aq1 = load_frag_a(Qrow + 32 + lhalf * 8);   // dk 32..63

  int qblk = qt >> 3;
  int ks = (qblk > 0) ? (qblk - 1) * 128 : 0;
  int ke = (qblk < 7) ? (qblk + 2) * 128 : 1024;

  const bf16* Kbase = Kb + (bh * 8200 + tbase) * 64;            // chunk keys
  const bf16* Vbase = Vtl + (bh * 8 + c) * 64 * (size_t)1024;   // chunk V^T

  float mrow[8], lsum[8];
#pragma unroll
  for (int j = 0; j < 8; ++j) { mrow[j] = -3.0e38f; lsum[j] = 0.0f; }
  f32x8 o[4];
#pragma unroll
  for (int nt = 0; nt < 4; ++nt) o[nt] = (f32x8)(0.0f);

  bf16* P = Plds[wv];
  const float scale = 0.125f;                       // 1/sqrt(64)

  auto loadK = [&](int kb, bf16x16* dst) {
    const bf16* Kr = Kbase + (size_t)(kb + lrow) * 64;
    dst[0] = load_frag_b(Kr + lhalf * 8);
    dst[1] = load_frag_b(Kr + 32 + lhalf * 8);
    dst[2] = load_frag_b(Kr + 1024 + lhalf * 8);        // +16 key rows
    dst[3] = load_frag_b(Kr + 1024 + 32 + lhalf * 8);
  };

  // One 32-key block: V loads first, next-K prefetch into `nk`, S-WMMAs from
  // `ck`, online softmax, P staging in LDS, P*V accumulation.
  auto step = [&](int kb, const bf16x16* ck, bf16x16* nk, int kbn) {
    bf16x16 vb[4];
#pragma unroll
    for (int nt = 0; nt < 4; ++nt)
      vb[nt] = load_frag_b(Vbase + (size_t)(nt * 16 + lrow) * 1024 + kb +
                           lhalf * 8);
    loadK(kbn, nk);

    f32x8 s0 = (f32x8)(0.0f), s1 = (f32x8)(0.0f);
    s0 = wmma_bf16(aq0, ck[0], s0);
    s0 = wmma_bf16(aq1, ck[1], s0);
    s1 = wmma_bf16(aq0, ck[2], s1);
    s1 = wmma_bf16(aq1, ck[3], s1);

    float p0[8], p1[8], corr[8];
#pragma unroll
    for (int j = 0; j < 8; ++j) {
      float a0 = s0[j] * scale, a1 = s1[j] * scale;
      float bm = rmax16(fmaxf(a0, a1));
      float mn = fmaxf(mrow[j], bm);
      corr[j] = __expf(mrow[j] - mn);
      p0[j] = __expf(a0 - mn);
      p1[j] = __expf(a1 - mn);
      lsum[j] = lsum[j] * corr[j] + rsum16(p0[j] + p1[j]);
      mrow[j] = mn;
    }
#pragma unroll
    for (int nt = 0; nt < 4; ++nt)
#pragma unroll
      for (int j = 0; j < 8; ++j) o[nt][j] *= corr[j];

#pragma unroll
    for (int j = 0; j < 8; ++j) {
      int row = j + 8 * lhalf;
      P[row * 32 + lrow] = (bf16)p0[j];
      P[row * 32 + 16 + lrow] = (bf16)p1[j];
    }
    bf16x16 ap = load_frag_a(P + lrow * 32 + lhalf * 8); // same-wave DS in-order

#pragma unroll
    for (int nt = 0; nt < 4; ++nt) o[nt] = wmma_bf16(ap, vb[nt], o[nt]);
  };

  bf16x16 kbuf0[4], kbuf1[4];
  loadK(ks, kbuf0);
  for (int kb = ks; kb < ke; kb += 64) {     // (ke-ks)/32 is 8 or 12: even
    step(kb, kbuf0, kbuf1, kb + 32);
    int kn = (kb + 64 < ke) ? (kb + 64) : (kb + 32);  // clamped final reload
    step(kb + 32, kbuf1, kbuf0, kn);
  }

  // ---- normalize, add chunk's global-token V row, store bf16 ----
#pragma unroll
  for (int nt = 0; nt < 4; ++nt) {
    float g = gV[(bh * 8 + c) * 64 + nt * 16 + lrow];
#pragma unroll
    for (int j = 0; j < 8; ++j) {
      int t = tbase + qt * 16 + j + 8 * lhalf;
      float v = o[nt][j] / lsum[j] + g;
      att[((size_t)b * 8200 + t) * 1024 + h * 64 + nt * 16 + lrow] = (bf16)v;
    }
  }
}

// ---------------------------------------------------------------------------
// Global-token attention (tiny): one 128-thread block per (b,h,c).
// ---------------------------------------------------------------------------
__global__ __launch_bounds__(128) void k_global(const bf16* __restrict__ Qb,
                                                const bf16* __restrict__ Kb,
                                                const bf16* __restrict__ Vtl,
                                                const float* __restrict__ gV,
                                                bf16* __restrict__ att) {
  __shared__ float sq[64];
  __shared__ float sc[1032];
  __shared__ float red[128];
  int tid = threadIdx.x;
  int gid = blockIdx.x;
  int c = gid & 7, h = (gid >> 3) & 15, b = gid >> 7;
  size_t bh = (size_t)b * 16 + h;

  const bf16* qrow = Qb + (bh * 8200 + (size_t)c * 1025) * 64;
  if (tid < 64) sq[tid] = (float)qrow[tid];
  __syncthreads();

  float lmax = -3.0e38f;
  for (int kk = tid; kk < 1032; kk += 128) {
    size_t t = (kk < 8) ? (size_t)kk * 1025 : ((size_t)c * 1025 + 1 + (kk - 8));
    const bf16* krow = Kb + (bh * 8200 + t) * 64;
    float s = 0.0f;
    for (int d = 0; d < 64; ++d) s += sq[d] * (float)krow[d];
    s *= 0.125f;
    sc[kk] = s;
    lmax = fmaxf(lmax, s);
  }
  red[tid] = lmax;
  __syncthreads();
  for (int s = 64; s > 0; s >>= 1) {
    if (tid < s) red[tid] = fmaxf(red[tid], red[tid + s]);
    __syncthreads();
  }
  float m = red[0];
  __syncthreads();

  float ls = 0.0f;
  for (int kk = tid; kk < 1032; kk += 128) {
    float p = __expf(sc[kk] - m);
    sc[kk] = p;
    ls += p;
  }
  red[tid] = ls;
  __syncthreads();
  for (int s = 64; s > 0; s >>= 1) {
    if (tid < s) red[tid] += red[tid + s];
    __syncthreads();
  }
  float l = red[0];

  if (tid < 64) {
    int d = tid;
    float acc = 0.0f;
    for (int e = 0; e < 8; ++e) acc += sc[e] * gV[(bh * 8 + e) * 64 + d];
    const bf16* vrow = Vtl + ((bh * 8 + c) * 64 + d) * (size_t)1024;
    for (int i = 0; i < 1024; ++i) acc += sc[8 + i] * (float)vrow[i];
    att[((size_t)b * 8200 + (size_t)c * 1025) * 1024 + h * 64 + d] =
        (bf16)(acc / l);
  }
}

// ---------------------------------------------------------------------------
// Output projection: att[16400x1024]bf16 x WoT[1024x1024]bf16 + bo -> f32.
// Same ping-pong 32x64-per-wave structure as k_qkv.
// ---------------------------------------------------------------------------
__global__ __launch_bounds__(128) void k_out(const bf16* __restrict__ A,
                                             const bf16* __restrict__ WoT,
                                             const float* __restrict__ bo,
                                             float* __restrict__ out) {
  int wave = blockIdx.x * 4 + (threadIdx.x >> 5);   // 0..8207
  int lane = threadIdx.x & 31;
  int lrow = lane & 15, lhalf = lane >> 4;
  int mgrp = wave >> 4;                             // 0..512
  int nstrip = wave & 15;
  int m0 = mgrp * 32, n0 = nstrip * 64;

  int r0 = m0 + lrow;
  int r1 = (m0 + 16 + lrow < 16400) ? (m0 + 16 + lrow) : r0;
  const bf16* arow0 = A + (size_t)r0 * 1024;
  const bf16* arow1 = A + (size_t)r1 * 1024;
  const bf16* brow[4];
#pragma unroll
  for (int nt = 0; nt < 4; ++nt)
    brow[nt] = WoT + (size_t)(n0 + nt * 16 + lrow) * 1024;

  f32x8 acc[8];
#pragma unroll
  for (int i = 0; i < 8; ++i) acc[i] = (f32x8)(0.0f);

  auto loadS = [&](int k, bf16x16& a0, bf16x16& a1, bf16x16* bb) {
    a0 = load_frag_a(arow0 + k + lhalf * 8);
    a1 = load_frag_a(arow1 + k + lhalf * 8);
#pragma unroll
    for (int nt = 0; nt < 4; ++nt)
      bb[nt] = load_frag_b(brow[nt] + k + lhalf * 8);
  };
  auto compS = [&](const bf16x16& a0, const bf16x16& a1, const bf16x16* bb) {
#pragma unroll
    for (int nt = 0; nt < 4; ++nt) {
      acc[nt * 2 + 0] = wmma_bf16(a0, bb[nt], acc[nt * 2 + 0]);
      acc[nt * 2 + 1] = wmma_bf16(a1, bb[nt], acc[nt * 2 + 1]);
    }
  };

  bf16x16 a0A, a1A, bA[4], a0B, a1B, bB[4];
  loadS(0, a0A, a1A, bA);
  for (int k = 32; k < 992; k += 64) {
    loadS(k, a0B, a1B, bB);
    compS(a0A, a1A, bA);
    loadS(k + 32, a0A, a1A, bA);
    compS(a0B, a1B, bB);
  }
  loadS(992, a0B, a1B, bB);
  compS(a0A, a1A, bA);
  compS(a0B, a1B, bB);

#pragma unroll
  for (int mi = 0; mi < 2; ++mi) {
    if (m0 + mi * 16 >= 16400) break;               // uniform tail guard
#pragma unroll
    for (int nt = 0; nt < 4; ++nt) {
      int ncol = n0 + nt * 16 + lrow;
      float bias = bo[ncol];
#pragma unroll
      for (int j = 0; j < 8; ++j) {
        int m = m0 + mi * 16 + j + 8 * lhalf;
        out[(size_t)m * 1024 + ncol] = acc[nt * 2 + mi][j] + bias;
      }
    }
  }
}

// ---------------------------------------------------------------------------
extern "C" void kernel_launch(void* const* d_in, const int* in_sizes, int n_in,
                              void* d_out, int out_size, void* d_ws,
                              size_t ws_size, hipStream_t stream) {
  (void)in_sizes; (void)n_in; (void)out_size; (void)ws_size;
  const float* x  = (const float*)d_in[0];
  const float* Wq = (const float*)d_in[1];
  const float* Wk = (const float*)d_in[2];
  const float* Wv = (const float*)d_in[3];
  const float* Wo = (const float*)d_in[4];
  const float* bo = (const float*)d_in[5];
  float* out = (float*)d_out;

  char* ws = (char*)d_ws;
  size_t off = 0;
  auto take = [&](size_t bytes) -> char* {
    char* p = ws + off;
    off += (bytes + 255) & ~(size_t)255;
    return p;
  };
  bf16* xb  = (bf16*)take((size_t)16400 * 1024 * 2);
  bf16* WqT = (bf16*)take((size_t)1024 * 1024 * 2);
  bf16* WkT = (bf16*)take((size_t)1024 * 1024 * 2);
  bf16* WvT = (bf16*)take((size_t)1024 * 1024 * 2);
  bf16* WoT = (bf16*)take((size_t)1024 * 1024 * 2);
  bf16* Qb  = (bf16*)take((size_t)2 * 16 * 8200 * 64 * 2);
  bf16* Kb  = (bf16*)take((size_t)2 * 16 * 8200 * 64 * 2);
  bf16* Vtl = (bf16*)take((size_t)2 * 16 * 8 * 64 * 1024 * 2);
  float* gV = (float*)take((size_t)2 * 16 * 8 * 64 * 4);
  bf16* att = (bf16*)take((size_t)16400 * 1024 * 2);

  k_cvt<<<8200, 256, 0, stream>>>(x, xb);           // 16.8M elems, 8/lane
  dim3 tgrid(32, 32), tblk(256);
  k_cvt_t<<<tgrid, tblk, 0, stream>>>(Wq, WqT);
  k_cvt_t<<<tgrid, tblk, 0, stream>>>(Wk, WkT);
  k_cvt_t<<<tgrid, tblk, 0, stream>>>(Wv, WvT);
  k_cvt_t<<<tgrid, tblk, 0, stream>>>(Wo, WoT);

  k_qkv<<<6156, 128, 0, stream>>>(xb, WqT, WkT, WvT, Qb, Kb, Vtl, gV);
  k_local<<<4096, 128, 0, stream>>>(Qb, Kb, Vtl, gV, att);
  k_global<<<256, 128, 0, stream>>>(Qb, Kb, Vtl, gV, att);
  k_out<<<2052, 128, 0, stream>>>(att, WoT, bo, out);
}
